// MaskPredictor_1949915152903
// MI455X (gfx1250) — matrix operand order repaired and checked by reading, not compiled, not run on hardware
//
#include <hip/hip_runtime.h>
#include <hip/hip_bf16.h>

// ---------------------------------------------------------------------------
// MaskPredictor for MI455X (gfx1250, wave32, WMMA).
// Pipeline: proj(q,k) -> kp^T@proj_n -> cheap_attn+softmax+top16 ->
//           basis threshold -> approx_attn (bf16 WMMA, async-LDS A tile) ->
//           top256 radix mask.
// ---------------------------------------------------------------------------

typedef __attribute__((ext_vector_type(16))) __bf16 v16bf;
typedef __attribute__((ext_vector_type(8)))  float  v8f;

#define Bb 8
#define Hh 12
#define Nn 1024
#define Cc 64
#define RC 32
#define RN 128
#define BH (Bb*Hh)

__device__ __forceinline__ v8f wmma_bf16(v16bf a, v16bf b, v8f c) {
    return __builtin_amdgcn_wmma_f32_16x16x32_bf16(
        /*neg_a=*/false, a, /*neg_b=*/false, b,
        /*c_mod=*/(short)0, c, /*reuse_a=*/false, /*reuse_b=*/false);
}

// A operand as packed here (16x32 bf16, MxK, contiguous k order): lanes 0-15 hold row M=lane with
// K in {0..15}; lanes 16-31 hold row M=lane-16 with K in {16..31}.
__device__ __forceinline__ v16bf load_a_f32g(const float* src, size_t sM, size_t sK,
                                             int m0, int k0, int mlimit) {
    const int lane = threadIdx.x & 31;
    const int m = m0 + (lane & 15);
    const int kh = (lane >> 4) * 16;
    const bool ok = (m < mlimit);
    v16bf a;
#pragma unroll
    for (int j = 0; j < 8; ++j) {
        int kk = k0 + kh + 2 * j;
        float x0 = ok ? src[(size_t)m * sM + (size_t)kk * sK] : 0.f;
        float x1 = ok ? src[(size_t)m * sM + (size_t)(kk + 1) * sK] : 0.f;
        a[2 * j]     = (__bf16)x0;
        a[2 * j + 1] = (__bf16)x1;
    }
    return a;
}

// B operand as packed here (32x16 bf16, KxN, contiguous k as A): VGPR j: lanes 0-15 hold K=2j,2j+1 at col=lane;
// lanes 16-31 hold K=16+2j,16+2j+1 at col=lane-16.
__device__ __forceinline__ v16bf load_b_f32g(const float* src, size_t sK, size_t sN,
                                             int k0, int n0) {
    const int lane = threadIdx.x & 31;
    const int n = n0 + (lane & 15);
    const int kh = (lane >> 4) * 16;
    v16bf b;
#pragma unroll
    for (int j = 0; j < 8; ++j) {
        int kk = k0 + kh + 2 * j;
        b[2 * j]     = (__bf16)src[(size_t)kk * sK + (size_t)n * sN];
        b[2 * j + 1] = (__bf16)src[(size_t)(kk + 1) * sK + (size_t)n * sN];
    }
    return b;
}

__device__ __forceinline__ v16bf load_b_bf16src(const __bf16* src, size_t sK, size_t sN,
                                                int k0, int n0) {
    const int lane = threadIdx.x & 31;
    const int n = n0 + (lane & 15);
    const int kh = (lane >> 4) * 16;
    v16bf b;
#pragma unroll
    for (int j = 0; j < 8; ++j) {
        int kk = k0 + kh + 2 * j;
        b[2 * j]     = src[(size_t)kk * sK + (size_t)n * sN];
        b[2 * j + 1] = src[(size_t)(kk + 1) * sK + (size_t)n * sN];
    }
    return b;
}

// C/D f32 16x16: VGPR r: lanes 0-15 -> (M=r, N=lane), lanes 16-31 -> (M=r+8).
__device__ __forceinline__ void store_cd(float* dst, size_t ld, int m0, int n0,
                                         v8f c, int mlimit) {
    const int lane = threadIdx.x & 31;
    const int n = n0 + (lane & 15);
    const int mo = (lane >> 4) * 8;
#pragma unroll
    for (int r2 = 0; r2 < 8; ++r2) {
        int m = m0 + mo + r2;
        if (m < mlimit) dst[(size_t)m * ld + n] = c[r2];
    }
}

// ---------------------------------------------------------------------------
// K1: qp = q @ w_q^T + b_q ; kp = k @ w_k^T + b_k   (one 16x16 tile per wave)
// ---------------------------------------------------------------------------
__global__ __launch_bounds__(256) void k1_proj(
        const float* __restrict__ q, const float* __restrict__ k,
        const float* __restrict__ w_q, const float* __restrict__ b_q,
        const float* __restrict__ w_k, const float* __restrict__ b_k,
        float* __restrict__ qp, float* __restrict__ kp) {
    int wid = blockIdx.x * 8 + (threadIdx.x >> 5);
    int nt = wid & 1;  wid >>= 1;
    int mt = wid & 63; wid >>= 6;
    int bh = wid % BH;
    int sel = wid / BH;
    const float* X    = sel ? k   : q;
    const float* W    = sel ? w_k : w_q;
    const float* bias = sel ? b_k : b_q;
    float*       Y    = sel ? kp  : qp;
    const float* xb = X + (size_t)bh * Nn * Cc;
    float*       yb = Y + (size_t)bh * Nn * RC;
    int m0 = mt * 16, n0 = nt * 16;
    const int lane = threadIdx.x & 31;
    float bv = bias[n0 + (lane & 15)];
    v8f c;
#pragma unroll
    for (int i = 0; i < 8; ++i) c[i] = bv;
#pragma unroll
    for (int k0 = 0; k0 < Cc; k0 += 32) {
        v16bf a  = load_a_f32g(xb, Cc, 1, m0, k0, 1 << 30);
        v16bf bm = load_b_f32g(W, 1, Cc, k0, n0);   // B[k=c][n=rc] = w[rc*C + c]
        c = wmma_bf16(a, bm, c);
    }
    store_cd(yb, RC, m0, n0, c, 1 << 30);
}

// ---------------------------------------------------------------------------
// K2: kp_proj[bh] = kp^T @ proj_n   ([32 x 1024] @ [1024 x 128])
// ---------------------------------------------------------------------------
__global__ __launch_bounds__(256) void k2_kproj(
        const float* __restrict__ kp, const float* __restrict__ proj_n,
        float* __restrict__ kpp) {
    int wid = blockIdx.x * 8 + (threadIdx.x >> 5);
    int nt = wid & 7; wid >>= 3;
    int mt = wid & 1; wid >>= 1;
    int bh = wid;
    const float* kb = kp + (size_t)bh * Nn * RC;
    float*       yb = kpp + (size_t)bh * RC * RN;
    int m0 = mt * 16, n0 = nt * 16;
    v8f c = {};
    for (int k0 = 0; k0 < Nn; k0 += 32) {
        v16bf a  = load_a_f32g(kb, 1, RC, m0, k0, 1 << 30);   // A[rc][n] = kp[n*RC+rc]
        v16bf bm = load_b_f32g(proj_n, RN, 1, k0, n0);
        c = wmma_bf16(a, bm, c);
    }
    store_cd(yb, RN, m0, n0, c, 1 << 30);
}

// ---------------------------------------------------------------------------
// K3: cheap_attn = qp @ kp_proj * scale -> softmax(RN) -> top-16 -> basis_coef
//     One block handles 16 rows x 128 cols; 8 waves do 8 WMMA N-tiles.
// ---------------------------------------------------------------------------
__global__ __launch_bounds__(256) void k3_cheap(
        const float* __restrict__ qp, const float* __restrict__ kpp,
        float* __restrict__ out_bc) {
    __shared__ float tile[16][136];
    __shared__ float rv[16][16];
    __shared__ int   ri[16][16];
    __shared__ float win_v[16][16];
    __shared__ int   win_i[16][16];
    __shared__ float rmax[16], rsum[16];

    int bid = blockIdx.x;
    int mt = bid & 63;
    int bh = bid >> 6;
    int tid = threadIdx.x;
    int wv = tid >> 5, lane = tid & 31;
    const float* qb = qp  + (size_t)bh * Nn * RC;
    const float* pb = kpp + (size_t)bh * RC * RN;
    int m0 = mt * 16, n0 = wv * 16;

    v8f c = {};
    v16bf a  = load_a_f32g(qb, RC, 1, m0, 0, 1 << 30);
    v16bf bm = load_b_f32g(pb, RN, 1, 0, n0);
    c = wmma_bf16(a, bm, c);

    const float scale = 0.28867513459481287f;   // 12^-0.5
    {
        int n = n0 + (lane & 15);
        int mo = (lane >> 4) * 8;
#pragma unroll
        for (int r2 = 0; r2 < 8; ++r2) tile[mo + r2][n] = c[r2] * scale;
    }
    __syncthreads();

    int r = tid >> 4, t = tid & 15;             // 16 threads per row
    // row softmax over 128
    float lm = -1e30f;
#pragma unroll
    for (int i = 0; i < 8; ++i) lm = fmaxf(lm, tile[r][t + 16 * i]);
    rv[r][t] = lm;
    __syncthreads();
    if (t == 0) { float m = -1e30f; for (int x = 0; x < 16; ++x) m = fmaxf(m, rv[r][x]); rmax[r] = m; }
    __syncthreads();
    float ls = 0.f;
#pragma unroll
    for (int i = 0; i < 8; ++i) {
        int j = t + 16 * i;
        float e = __expf(tile[r][j] - rmax[r]);
        tile[r][j] = e; ls += e;
    }
    rv[r][t] = ls;
    __syncthreads();
    if (t == 0) { float s = 0.f; for (int x = 0; x < 16; ++x) s += rv[r][x]; rsum[r] = s; }
    __syncthreads();
    float inv = 1.f / rsum[r];
#pragma unroll
    for (int i = 0; i < 8; ++i) tile[r][t + 16 * i] *= inv;
    __syncthreads();

    // iterative top-16 (16 uniform iterations across the block)
    for (int it = 0; it < 16; ++it) {
        float bv = -1e30f; int bi = 0;
#pragma unroll
        for (int i = 0; i < 8; ++i) {
            int j = t + 16 * i;
            float v = tile[r][j];
            if (v > bv) { bv = v; bi = j; }
        }
        rv[r][t] = bv; ri[r][t] = bi;
        __syncthreads();
        if (t == 0) {
            float gb = -1e30f; int gi = 0;
            for (int x = 0; x < 16; ++x)
                if (rv[r][x] > gb) { gb = rv[r][x]; gi = ri[r][x]; }
            win_v[r][it] = gb; win_i[r][it] = gi;
            tile[r][gi] = -1e30f;
        }
        __syncthreads();
    }

    // scatter: zero row, set 16 winners (row 0 of the full N is the dropped cls)
    int m_global = m0 + r;
    if (m_global >= 1) {
        float* orow = out_bc + ((size_t)bh * (Nn - 1) + (size_t)(m_global - 1)) * RN;
#pragma unroll
        for (int i = 0; i < 8; ++i) {
            int j = t + 16 * i;
            float v = 0.f;
            for (int w = 0; w < 16; ++w)
                if (win_i[r][w] == j) v = win_v[r][w];
            orow[j] = v;
        }
    }
}

// ---------------------------------------------------------------------------
// K4: basis[r][n] = |proj_back_n[n][r]| thresholded, stored bf16 for WMMA B.
// ---------------------------------------------------------------------------
__global__ __launch_bounds__(256) void k4_basis(
        const float* __restrict__ pbn, __bf16* __restrict__ basisbf) {
    int idx = blockIdx.x * 256 + threadIdx.x;
    if (idx >= RN * Nn) return;
    int rr = idx >> 10, n = idx & (Nn - 1);
    float v = fabsf(pbn[(size_t)n * RN + rr]);
    basisbf[idx] = (__bf16)((v > 0.02f) ? v : 0.f);
}

// ---------------------------------------------------------------------------
// K5: approx_attn = basis_coef @ basis   ([1023 x 128] @ [128 x 1024] per bh)
//     Block = 8 waves sharing one 16x128 A tile, staged into LDS with the
//     gfx1250 async-to-LDS DMA path (GLOBAL_LOAD_ASYNC_TO_LDS_B128, ASYNCcnt).
// ---------------------------------------------------------------------------
__global__ __launch_bounds__(256) void k5_approx(
        const float* __restrict__ bc, const __bf16* __restrict__ basisbf,
        float* __restrict__ aa) {
    __shared__ float As[16][132];               // 528-byte rows, 16B aligned
    int bid = blockIdx.x;
    int ntg = bid & 7;
    int mt  = (bid >> 3) & 63;
    int bh  = bid >> 9;
    int tid = threadIdx.x;
    const float* ab = bc + (size_t)bh * (Nn - 1) * RN;
    float*       yb = aa + (size_t)bh * (Nn - 1) * Nn;
    int m0 = mt * 16;

    // Async-stage A tile: 16 rows x 512B = 512 x b128 transfers, 2 per thread.
    // VDST holds the LDS byte address (low 32 bits of the flat shared pointer,
    // per the LDS aperture mapping); SADDR holds the uniform global base.
    uint32_t lds_base = (uint32_t)(uintptr_t)&As[0][0];
    for (int e = tid; e < 512; e += 256) {
        int rr = e >> 5, seg = e & 31;          // row, 16B segment within row
        int m = m0 + rr;
        uint32_t loff = lds_base + (uint32_t)(rr * 528 + seg * 16);
        if (m < Nn - 1) {
            uint32_t goff = (uint32_t)(((size_t)m * RN + (size_t)seg * 4) * sizeof(float));
            asm volatile("global_load_async_to_lds_b128 %0, %1, %2"
                         :: "v"(loff), "v"(goff), "s"(ab)
                         : "memory");
        } else {                                // zero-fill the pad row
            As[rr][seg * 4 + 0] = 0.f;
            As[rr][seg * 4 + 1] = 0.f;
            As[rr][seg * 4 + 2] = 0.f;
            As[rr][seg * 4 + 3] = 0.f;
        }
    }
    if (m0 + 16 < Nn - 1)
        __builtin_prefetch(ab + (size_t)(m0 + 16) * RN, 0, 1);  // global_prefetch_b8
    asm volatile("s_wait_asynccnt 0x0" ::: "memory");
    __syncthreads();

    int wv = tid >> 5;
    int n0 = (ntg * 8 + wv) * 16;
    v8f c = {};
#pragma unroll
    for (int ks = 0; ks < 4; ++ks) {
        v16bf a  = load_a_f32g(&As[0][0], 132, 1, 0, ks * 32, 1 << 30);
        v16bf bm = load_b_bf16src(basisbf, Nn, 1, ks * 32, n0);
        c = wmma_bf16(a, bm, c);
    }
    store_cd(yb, Nn, m0, n0, c, Nn - 1);
}

// ---------------------------------------------------------------------------
// K6: attn_mask — exact top-256-of-1024 binary mask via 4x8-bit radix select.
//     One 256-thread block per output row (4 values/thread in registers).
// ---------------------------------------------------------------------------
__global__ __launch_bounds__(256) void k6_mask(
        const float* __restrict__ aa, float* __restrict__ mask) {
    __shared__ int hist[256];
    __shared__ unsigned s_prefix;
    __shared__ int s_kk;
    __shared__ int tiecnt;

    int bid = blockIdx.x;
    int rr = bid & (Nn - 1);
    int bh = bid >> 10;
    int tid = threadIdx.x;
    float* orow = mask + ((size_t)bh * Nn + rr) * Nn;

    if (rr == 0) {                         // cls row: all ones
#pragma unroll
        for (int i = 0; i < 4; ++i) orow[tid + 256 * i] = 1.0f;
        return;
    }
    const float* srow = aa + ((size_t)bh * (Nn - 1) + (rr - 1)) * Nn;

    unsigned u[4];                         // order-preserving uint keys
#pragma unroll
    for (int i = 0; i < 4; ++i) {
        unsigned bits = __float_as_uint(srow[tid + 256 * i]);
        u[i] = bits ^ ((unsigned)((int)bits >> 31) | 0x80000000u);
    }

    unsigned prefix = 0;
    int kk = 256;
    for (int pass = 3; pass >= 0; --pass) {
        hist[tid] = 0;
        __syncthreads();
        int shift = pass * 8;
        unsigned hm = (pass == 3) ? 0u : (0xFFFFFFFFu << (shift + 8));
#pragma unroll
        for (int i = 0; i < 4; ++i)
            if ((u[i] & hm) == (prefix & hm))
                atomicAdd(&hist[(u[i] >> shift) & 0xFF], 1);
        __syncthreads();
        if (tid == 0) {
            int rem = kk, b = 255;
            for (; b > 0; --b) { int c = hist[b]; if (rem - c <= 0) break; rem -= c; }
            s_prefix = prefix | ((unsigned)b << shift);
            s_kk = rem;
        }
        __syncthreads();
        prefix = s_prefix; kk = s_kk;
        __syncthreads();
    }

    if (tid == 0) tiecnt = 0;
    __syncthreads();
    unsigned Tu = prefix;                  // exact value of 256th-largest key
#pragma unroll
    for (int i = 0; i < 4; ++i) {
        float o;
        if (u[i] > Tu) o = 1.0f;
        else if (u[i] == Tu) { int p = atomicAdd(&tiecnt, 1); o = (p < kk) ? 1.0f : 0.0f; }
        else o = 0.0f;
        orow[tid + 256 * i] = o;
    }
}

// ---------------------------------------------------------------------------
extern "C" void kernel_launch(void* const* d_in, const int* in_sizes, int n_in,
                              void* d_out, int out_size, void* d_ws, size_t ws_size,
                              hipStream_t stream) {
    (void)in_sizes; (void)n_in; (void)out_size; (void)ws_size;
    const float* q      = (const float*)d_in[0];
    const float* k      = (const float*)d_in[1];
    const float* w_q    = (const float*)d_in[2];
    const float* b_q    = (const float*)d_in[3];
    const float* w_k    = (const float*)d_in[4];
    const float* b_k    = (const float*)d_in[5];
    const float* proj_n = (const float*)d_in[6];
    const float* pbn    = (const float*)d_in[7];

    // workspace partition (~27 MB)
    float* qp  = (float*)d_ws;
    float* kp  = qp + (size_t)BH * Nn * RC;
    float* kpp = kp + (size_t)BH * Nn * RC;
    __bf16* basisbf = (__bf16*)(kpp + (size_t)BH * RC * RN);

    // outputs, concatenated flat in reference return order
    float* out_bc   = (float*)d_out;                                  // [BH,1023,128]
    float* out_aa   = out_bc + (size_t)BH * (Nn - 1) * RN;            // [BH,1023,1024]
    float* out_mask = out_aa + (size_t)BH * (Nn - 1) * Nn;            // [BH,1024,1024]

    k1_proj <<<(2 * BH * 64 * 2) / 8, 256, 0, stream>>>(q, k, w_q, b_q, w_k, b_k, qp, kp);
    k2_kproj<<<(BH * 2 * 8) / 8,     256, 0, stream>>>(kp, proj_n, kpp);
    k3_cheap<<<BH * 64,              256, 0, stream>>>(qp, kpp, out_bc);
    k4_basis<<<(RN * Nn) / 256,      256, 0, stream>>>(pbn, basisbf);
    k5_approx<<<BH * 64 * 8,         256, 0, stream>>>(out_bc, basisbf, out_aa);
    k6_mask <<<BH * Nn,              256, 0, stream>>>(out_aa, out_mask);
}
